// LatentRecurrent_83726092468349
// MI455X (gfx1250) — compile-verified, matrix operand order checked by reading, not run on hardware
//
#include <hip/hip_runtime.h>
#include <hip/hip_bf16.h>

// ---------------------------------------------------------------------------
// LatentRecurrent for MI455X (gfx1250, wave32, WMMA).
//
// out[b,t,:] = z_t[b,:] - z_0[b,:], with
//   temp = [lrelu(z W1^T + b1), h]           (bf16 staged in LDS)
//   z_{t+1} = lrelu(temp W2^T + b2)*r + z_t  (fp32 residual chain via out)
//   h_{t+1} = lrelu(temp W3^T + b3)*r + h_t
//
// Block = 128 threads (4 waves) owning 64 rows. Waves split the N dimension;
// each wave applies one B-fragment (from L2-resident bf16 weights) to 4
// A-fragments / 4 accumulators -> 4x B-reuse, 4 concurrent WMMA chains.
// ---------------------------------------------------------------------------

typedef __bf16 bf16_t;
typedef __attribute__((ext_vector_type(16))) __bf16 v16bf;
typedef __attribute__((ext_vector_type(8)))  __bf16 v8bf;
typedef __attribute__((ext_vector_type(8)))  float  v8f;

union FragAB { v16bf v; v8bf h[2]; };

// 16x32 bf16 A/B fragment: lane owns K = {0..7,16..23} (lanes 0-15) or
// {8..15,24..31} (lanes 16-31); p points at the lane's first K element.
__device__ inline v16bf load_frag(const bf16_t* p) {
    FragAB f;
    f.h[0] = *(const v8bf*)(p);
    f.h[1] = *(const v8bf*)(p + 16);
    return f.v;
}

__device__ inline float lrelu(float x) { return x > 0.0f ? x : 0.01f * x; }

#define B_DIM   4096
#define Z_DIM   512
#define H_DIM   384
#define H2_DIM  768
#define T_DIM   32       // n_frames (harness constant; device scalar is
                         // unreadable during graph capture)
#define MT      64       // rows per block (4 M-tiles of 16)
#define NWAVES  4
#define ZP      520      // LDS stride, z tile (16B aligned, 4-dword bank shift)
#define TP      776      // LDS stride, temp tile
#define SMEM_BYTES (MT*ZP*2 + MT*TP*2)   // 66560 + 99328 = 165888 B

struct Acc4 { v8f a[4]; };

// Accumulate 4 M-tiles x 1 N-tile over K: one B-frag load feeds 4 WMMAs.
__device__ inline Acc4 gemm_tiles(const bf16_t* __restrict__ sA, int sstride,
                                  const bf16_t* __restrict__ brow, int kblocks,
                                  int lidx, int half)
{
    Acc4 r;
#pragma unroll
    for (int mt = 0; mt < 4; ++mt)
        r.a[mt] = (v8f){0.f, 0.f, 0.f, 0.f, 0.f, 0.f, 0.f, 0.f};
    const bf16_t* aBase = sA + lidx * sstride + half * 8;
#pragma unroll 2
    for (int kb = 0; kb < kblocks; ++kb) {
        v16bf b = load_frag(brow + kb * 32);
#pragma unroll
        for (int mt = 0; mt < 4; ++mt) {
            v16bf a = load_frag(aBase + (mt * 16) * sstride + kb * 32);
            r.a[mt] = __builtin_amdgcn_wmma_f32_16x16x32_bf16(
                false, a, false, b, (short)0, r.a[mt], false, false);
        }
    }
    return r;
}

__global__ void __launch_bounds__(128)
lr_step_kernel(const float* __restrict__ z0,
               float* __restrict__ out,            // [B, T, Z] deltas
               const float* __restrict__ hprev,    // [B, H] fp32
               float* __restrict__ hnext,          // [B, H] fp32
               const bf16_t* __restrict__ w1,      // [H, Z]  bf16
               const float*  __restrict__ b1,
               const bf16_t* __restrict__ w2,      // [Z, 2H] bf16
               const float*  __restrict__ b2,
               const bf16_t* __restrict__ w3,      // [H, 2H] bf16
               const float*  __restrict__ b3,
               int t, float r)
{
    extern __shared__ __align__(16) char smem[];
    bf16_t* sZ = (bf16_t*)smem;                    // [MT][ZP]  z_prev (bf16)
    bf16_t* sT = (bf16_t*)(smem + MT * ZP * 2);    // [MT][TP]  temp = [t1 | h]

    const int tid  = threadIdx.x;        // 0..127
    const int lane = tid & 31;
    const int wid  = tid >> 5;           // wave 0..3 -> N-tiles wid::4
    const int half = lane >> 4;          // K-half selector
    const int lidx = lane & 15;          // row (A) / col (B,D) within tile
    const int row0 = blockIdx.x * MT;    // first global row of this block

    // ---- Phase 0: stage z_prev and h_prev tiles into LDS as bf16 ----------
    // z_prev[b,c] = z0[b,c] + (t>1 ? out[b,t-1,c] : 0)
    for (int i = tid; i < MT * (Z_DIM / 4); i += 128) {
        const int rr = i / (Z_DIM / 4);
        const int cc = (i % (Z_DIM / 4)) * 4;
        const size_t gr = (size_t)(row0 + rr);
        float4 z4 = *(const float4*)(z0 + gr * Z_DIM + cc);
        if (t > 1) {
            float4 d4 = *(const float4*)(out + (gr * T_DIM + (t - 1)) * Z_DIM + cc);
            z4.x += d4.x; z4.y += d4.y; z4.z += d4.z; z4.w += d4.w;
        }
        bf16_t* dst = sZ + rr * ZP + cc;
        dst[0] = (bf16_t)z4.x; dst[1] = (bf16_t)z4.y;
        dst[2] = (bf16_t)z4.z; dst[3] = (bf16_t)z4.w;
    }
    for (int i = tid; i < MT * (H_DIM / 4); i += 128) {
        const int rr = i / (H_DIM / 4);
        const int cc = (i % (H_DIM / 4)) * 4;
        float4 h4 = *(const float4*)(hprev + (size_t)(row0 + rr) * H_DIM + cc);
        bf16_t* dst = sT + rr * TP + H_DIM + cc;   // h occupies cols [H, 2H)
        dst[0] = (bf16_t)h4.x; dst[1] = (bf16_t)h4.y;
        dst[2] = (bf16_t)h4.z; dst[3] = (bf16_t)h4.w;
    }
    __syncthreads();

    // ---- Phase 1 (GEMM1): t1 = lrelu(z W1^T + b1) -> sT cols [0, H) -------
    for (int nb = wid; nb < H_DIM / 16; nb += NWAVES) {
        const int n = nb * 16 + lidx;
        Acc4 acc = gemm_tiles(sZ, ZP, w1 + (size_t)n * Z_DIM + half * 8,
                              Z_DIM / 32, lidx, half);
        const float bias = b1[n];
#pragma unroll
        for (int mt = 0; mt < 4; ++mt) {
#pragma unroll
            for (int v = 0; v < 8; ++v) {
                const int m = mt * 16 + half * 8 + v;
                sT[m * TP + n] = (bf16_t)lrelu(acc.a[mt][v] + bias);
            }
        }
    }
    __syncthreads();

    // ---- Phase 2 (GEMM2): out[b,t,n] = lrelu(temp W2^T + b2)*r + out[b,t-1,n]
    for (int nb = wid; nb < Z_DIM / 16; nb += NWAVES) {
        const int n = nb * 16 + lidx;
        Acc4 acc = gemm_tiles(sT, TP, w2 + (size_t)n * H2_DIM + half * 8,
                              H2_DIM / 32, lidx, half);
        const float bias = b2[n];
#pragma unroll
        for (int mt = 0; mt < 4; ++mt) {
#pragma unroll
            for (int v = 0; v < 8; ++v) {
                const int m = mt * 16 + half * 8 + v;
                const size_t gr = (size_t)(row0 + m);
                const float x = lrelu(acc.a[mt][v] + bias);
                const float prev =
                    (t > 1) ? out[(gr * T_DIM + (t - 1)) * Z_DIM + n] : 0.0f;
                out[(gr * T_DIM + t) * Z_DIM + n] = fmaf(x, r, prev);
            }
        }
    }

    // ---- Phase 3 (GEMM3): hnext = lrelu(temp W3^T + b3)*r + hprev ---------
    for (int nb = wid; nb < H_DIM / 16; nb += NWAVES) {
        const int n = nb * 16 + lidx;
        Acc4 acc = gemm_tiles(sT, TP, w3 + (size_t)n * H2_DIM + half * 8,
                              H2_DIM / 32, lidx, half);
        const float bias = b3[n];
#pragma unroll
        for (int mt = 0; mt < 4; ++mt) {
#pragma unroll
            for (int v = 0; v < 8; ++v) {
                const int m = mt * 16 + half * 8 + v;
                const size_t gr = (size_t)(row0 + m);
                const float x = lrelu(acc.a[mt][v] + bias);
                hnext[gr * H_DIM + n] = fmaf(x, r, hprev[gr * H_DIM + n]);
            }
        }
    }
}

// fp32 -> bf16 weight conversion (rerun every call; no cached state).
__global__ void cvt_bf16_kernel(const float* __restrict__ src,
                                bf16_t* __restrict__ dst, int n)
{
    int i = blockIdx.x * blockDim.x + threadIdx.x;
    if (i < n) dst[i] = (bf16_t)src[i];
}

// out[:, 0, :] = 0  (frame 0 delta; d_out is poisoned by harness).
__global__ void init_out_kernel(float* __restrict__ out)
{
    int i = blockIdx.x * blockDim.x + threadIdx.x;   // over B * Z/4
    const int b = i / (Z_DIM / 4);
    const int c = (i % (Z_DIM / 4)) * 4;
    float4 z = {0.f, 0.f, 0.f, 0.f};
    *(float4*)(out + (size_t)b * T_DIM * Z_DIM + c) = z;
}

extern "C" void kernel_launch(void* const* d_in, const int* in_sizes, int n_in,
                              void* d_out, int out_size, void* d_ws, size_t ws_size,
                              hipStream_t stream) {
    const float* z0     = (const float*)d_in[0];
    const float* h_init = (const float*)d_in[1];
    const float* W1     = (const float*)d_in[2];
    const float* b1     = (const float*)d_in[3];
    const float* W2     = (const float*)d_in[4];
    const float* b2     = (const float*)d_in[5];
    const float* W3     = (const float*)d_in[6];
    const float* b3     = (const float*)d_in[7];
    float* out = (float*)d_out;

    // Workspace layout (all 16B aligned):
    //   [0,        393216)  W1 bf16  (384*512)
    //   [393216,  1179648)  W2 bf16  (512*768)
    //   [1179648, 1769472)  W3 bf16  (384*768)
    //   [1769472, 8060928)  h ping   (4096*384 fp32)
    //   [8060928,14352384)  h pong
    char* ws = (char*)d_ws;
    bf16_t* w1bf = (bf16_t*)(ws);
    bf16_t* w2bf = (bf16_t*)(ws + 393216);
    bf16_t* w3bf = (bf16_t*)(ws + 1179648);
    float*  hbA  = (float*)(ws + 1769472);
    float*  hbB  = (float*)(ws + 1769472 + 6291456);

    const int nW1 = H_DIM * Z_DIM;      // 196608
    const int nW2 = Z_DIM * H2_DIM;     // 393216
    const int nW3 = H_DIM * H2_DIM;     // 294912
    cvt_bf16_kernel<<<(nW1 + 255) / 256, 256, 0, stream>>>(W1, w1bf, nW1);
    cvt_bf16_kernel<<<(nW2 + 255) / 256, 256, 0, stream>>>(W2, w2bf, nW2);
    cvt_bf16_kernel<<<(nW3 + 255) / 256, 256, 0, stream>>>(W3, w3bf, nW3);

    init_out_kernel<<<(B_DIM * (Z_DIM / 4)) / 256, 256, 0, stream>>>(out);

    const float r = 0.2f * 4.0f / (float)T_DIM;   // 0.025

    for (int t = 1; t < T_DIM; ++t) {
        const float* hp = (t == 1) ? h_init : (((t - 1) & 1) ? hbA : hbB);
        float*       hn = (t & 1) ? hbA : hbB;
        lr_step_kernel<<<B_DIM / MT, 128, SMEM_BYTES, stream>>>(
            z0, out, hp, hn, w1bf, b1, w2bf, b2, w3bf, b3, t, r);
    }
}
